// LSTM_44538810860368
// MI455X (gfx1250) — compile-verified, matrix operand order checked by reading, not run on hardware
//
#include <hip/hip_runtime.h>
#include <hip/hip_bf16.h>
#include <math.h>

// ---------------------------------------------------------------------------
// LSTM on MI455X (gfx1250, wave32, WMMA)
//   B=256, T=512, E=256, H=1024, 4H=4096, VOCAB=3, NCLS=10
// VOCAB=3 -> input projection collapses to a [3][4096] lookup table.
// The sequential recurrence (512 x [256x1024]@[1024x4096]) runs on
// v_wmma_f32_16x16x32_f16 with wh pre-packed into B-fragment lane layout.
// k-loop is software-pipelined (register double-buffering); this revision
// relaxes the occupancy target (launch_bounds min-waves=1: grid, not regs,
// bounds occupancy here) and adds sched_group_barrier hints so next-chunk
// loads are scheduled ahead of the current chunk's 4 WMMAs.
// ---------------------------------------------------------------------------

typedef _Float16 v16h __attribute__((ext_vector_type(16)));
typedef _Float16 v8h  __attribute__((ext_vector_type(8)));
typedef float    v8f  __attribute__((ext_vector_type(8)));

#define BATCH 256
#define SEQ   512
#define EMBD  256
#define HID   1024
#define G4    4096
#define NCLS  10

// K-offset within a 32-wide chunk for f16 fragment slot s on a given lane
// half (ISA 7.12.2, 16-bit A 16x32 layout; B symmetric column form):
// lanes 0-15 hold K {0..7, 16..23}; lanes 16-31 hold K {8..15, 24..31}.
__device__ __forceinline__ int frag_koff(int s, bool hi) {
    int p = s >> 1, w = s & 1;
    int base = (p < 4) ? (2 * p) : (16 + 2 * (p - 4));
    return base + w + (hi ? 8 : 0);
}

// ---------------------------------------------------------------------------
// K0: pack wh = [w_gh | w_ih | w_fh | w_oh] ([1024][4096] f32) into f16
// fragments: whp[((nt*32)+kc)*512 + lane*16 + s]
// ---------------------------------------------------------------------------
__global__ __launch_bounds__(256) void pack_wh_kernel(
    const float* __restrict__ w_gh, const float* __restrict__ w_ih,
    const float* __restrict__ w_fh, const float* __restrict__ w_oh,
    _Float16* __restrict__ whp)
{
    int tid  = blockIdx.x * 256 + threadIdx.x;   // 0..262143
    int lane = tid & 31;
    int frag = tid >> 5;                          // 0..8191
    int kc   = frag & 31;
    int nt   = frag >> 5;                         // 0..255
    int n    = nt * 16 + (lane & 15);             // global column in [0,4096)
    int gate = n >> 10;
    int col  = n & 1023;
    const float* w = (gate == 0) ? w_gh : (gate == 1) ? w_ih
                   : (gate == 2) ? w_fh : w_oh;
    bool hi = lane >= 16;
    int  k0 = kc * 32;
    _Float16* dst = whp + (size_t)frag * 512 + lane * 16;
#pragma unroll
    for (int s = 0; s < 16; ++s) {
        int k = k0 + frag_koff(s, hi);
        dst[s] = (_Float16)w[(size_t)k * HID + col];
    }
}

// ---------------------------------------------------------------------------
// K1: gate LUT  xtab[v][n] = b[n] + sum_e emb[v][e] * wx[e][n]   (3 x 4096)
// ---------------------------------------------------------------------------
__global__ __launch_bounds__(256) void gate_table_kernel(
    const float* __restrict__ emb,
    const float* __restrict__ w_gx, const float* __restrict__ w_ix,
    const float* __restrict__ w_fx, const float* __restrict__ w_ox,
    const float* __restrict__ b_g,  const float* __restrict__ b_i,
    const float* __restrict__ b_f,  const float* __restrict__ b_o,
    float* __restrict__ xtab)
{
    int tid = blockIdx.x * 256 + threadIdx.x;
    if (tid >= 3 * G4) return;
    int v    = tid / G4;
    int n    = tid % G4;
    int gate = n >> 10;
    int col  = n & 1023;
    const float* w  = (gate == 0) ? w_gx : (gate == 1) ? w_ix
                    : (gate == 2) ? w_fx : w_ox;
    const float* bb = (gate == 0) ? b_g  : (gate == 1) ? b_i
                    : (gate == 2) ? b_f  : b_o;
    float s = bb[col];
    const float* ev = emb + v * EMBD;
    for (int e = 0; e < EMBD; ++e)
        s += ev[e] * w[(size_t)e * HID + col];
    xtab[tid] = s;
}

// ---------------------------------------------------------------------------
// K2: one recurrence step.
//   grid = 128 blocks x 256 thr (8 waves). Wave owns one 16x16 h-tile;
//   all waves in a block share j_tile (B fragments hit in WGP$).
//   Software-pipelined k-loop; min-waves-per-EU = 1 so the compiler may
//   spend VGPRs on keeping both fragment buffers live (grid-bound occupancy).
// ---------------------------------------------------------------------------
__global__ __launch_bounds__(256, 1) void lstm_step_kernel(
    const _Float16* __restrict__ h_in,   // [256][1024] f16
    _Float16*       __restrict__ h_out,  // [256][1024] f16
    float*          __restrict__ cst,    // [256][1024] f32 (in place)
    const _Float16* __restrict__ whp,    // packed wh fragments
    const float*    __restrict__ xtab,   // [3][4096]
    const int*      __restrict__ x,      // [256][512]
    int t)
{
    const int lane   = threadIdx.x & 31;
    const int wave   = threadIdx.x >> 5;       // 0..7
    const int j_tile = blockIdx.x >> 1;        // 0..63 (hidden col tile)
    const int m_half = blockIdx.x & 1;
    const int m_tile = m_half * 8 + wave;      // 0..15 (batch row tile)
    const int m0 = m_tile * 16;
    const int j0 = j_tile * 16;
    const bool hi  = lane >= 16;
    const int  l15 = lane & 15;
    const int  rowA = m0 + l15;
    const int  aoff = hi ? 8 : 0;

    v8f acc0 = {}, acc1 = {}, acc2 = {}, acc3 = {};

    const _Float16* arow = h_in + (size_t)rowA * HID + aoff;
    const size_t gate_stride = (size_t)64 * 32 * 512;   // per-gate fragment stride
    const _Float16* bp0 = whp + (size_t)j_tile * 32 * 512 + lane * 16;
    const _Float16* bp1 = bp0 + gate_stride;
    const _Float16* bp2 = bp1 + gate_stride;
    const _Float16* bp3 = bp2 + gate_stride;

    // Double-buffered fragment registers.
    v8h  a0[2], a1[2];
    v16h b0[2], b1[2], b2[2], b3[2];

    // Prologue: load chunk 0.
    a0[0] = *(const v8h*)(arow);
    a1[0] = *(const v8h*)(arow + 16);
    b0[0] = *(const v16h*)(bp0);
    b1[0] = *(const v16h*)(bp1);
    b2[0] = *(const v16h*)(bp2);
    b3[0] = *(const v16h*)(bp3);

#pragma unroll
    for (int kc = 0; kc < 32; ++kc) {
        const int cur = kc & 1;
        const int nxt = cur ^ 1;
        // Prefetch next chunk before consuming the current one: these loads
        // stay in flight across the 4 WMMAs below (partial loadcnt waits).
        if (kc + 1 < 32) {
            const _Float16* an = arow + (kc + 1) * 32;
            a0[nxt] = *(const v8h*)(an);
            a1[nxt] = *(const v8h*)(an + 16);
            b0[nxt] = *(const v16h*)(bp0 + (size_t)(kc + 1) * 512);
            b1[nxt] = *(const v16h*)(bp1 + (size_t)(kc + 1) * 512);
            b2[nxt] = *(const v16h*)(bp2 + (size_t)(kc + 1) * 512);
            b3[nxt] = *(const v16h*)(bp3 + (size_t)(kc + 1) * 512);
        }
        v16h a;
#pragma unroll
        for (int i = 0; i < 8; ++i) { a[i] = a0[cur][i]; a[8 + i] = a1[cur][i]; }

        acc0 = __builtin_amdgcn_wmma_f32_16x16x32_f16(false, a, false, b0[cur],
                                                      (short)0, acc0, false, false);
        acc1 = __builtin_amdgcn_wmma_f32_16x16x32_f16(false, a, false, b1[cur],
                                                      (short)0, acc1, false, false);
        acc2 = __builtin_amdgcn_wmma_f32_16x16x32_f16(false, a, false, b2[cur],
                                                      (short)0, acc2, false, false);
        acc3 = __builtin_amdgcn_wmma_f32_16x16x32_f16(false, a, false, b3[cur],
                                                      (short)0, acc3, false, false);

        // Scheduling hint: per iteration, group the 10 VMEM reads (next
        // chunk's fragments) ahead of the 4 matrix ops (current chunk).
        // Degrades to a no-op if the pass ignores WMMA on this target.
        __builtin_amdgcn_sched_group_barrier(0x020, 10, 0);  // VMEM read x10
        __builtin_amdgcn_sched_group_barrier(0x008, 4, 0);   // MFMA/WMMA x4
    }

    // Epilogue. C layout: VGPR v -> row m0 + v (+8 for hi half); lane -> col.
    const int j = j0 + l15;
#pragma unroll
    for (int v = 0; v < 8; ++v) {
        int m = m0 + v + (hi ? 8 : 0);
        int tok = x[m * SEQ + t];
        const float* xt = xtab + (size_t)tok * G4 + j;
        float pg = acc0[v] + xt[0 * HID];
        float pi = acc1[v] + xt[1 * HID];
        float pf = acc2[v] + xt[2 * HID];
        float po = acc3[v] + xt[3 * HID];
        float gv = tanhf(pg);
        float iv = 1.0f / (1.0f + __expf(-pi));
        float fv = 1.0f / (1.0f + __expf(-pf));
        float ov = 1.0f / (1.0f + __expf(-po));
        size_t idx = (size_t)m * HID + j;
        float cn = gv * iv + cst[idx] * fv;
        cst[idx] = cn;
        h_out[idx] = (_Float16)(tanhf(cn) * ov);
    }
}

// ---------------------------------------------------------------------------
// K3: p = h @ w_ph + b_p ; out = log_softmax(p, axis=0)  (over batch!)
// ---------------------------------------------------------------------------
__global__ __launch_bounds__(256) void proj_logsoftmax_kernel(
    const _Float16* __restrict__ h,     // [256][1024] f16 (final h)
    const float* __restrict__ w_ph,     // [1024][10]
    const float* __restrict__ b_p,      // [10]
    float* __restrict__ out)            // [256][10]
{
    __shared__ float sp[BATCH][NCLS];
    __shared__ float slse[NCLS];
    int b = threadIdx.x;

    float acc[NCLS];
#pragma unroll
    for (int n = 0; n < NCLS; ++n) acc[n] = b_p[n];
    for (int j = 0; j < HID; ++j) {
        float hv = (float)h[(size_t)b * HID + j];
#pragma unroll
        for (int n = 0; n < NCLS; ++n) acc[n] += hv * w_ph[j * NCLS + n];
    }
#pragma unroll
    for (int n = 0; n < NCLS; ++n) sp[b][n] = acc[n];
    __syncthreads();

    if (b < NCLS) {
        float mx = -INFINITY;
        for (int r = 0; r < BATCH; ++r) mx = fmaxf(mx, sp[r][b]);
        float s = 0.0f;
        for (int r = 0; r < BATCH; ++r) s += __expf(sp[r][b] - mx);
        slse[b] = mx + __logf(s);
    }
    __syncthreads();

#pragma unroll
    for (int n = 0; n < NCLS; ++n)
        out[b * NCLS + n] = sp[b][n] - slse[n];
}

// ---------------------------------------------------------------------------
extern "C" void kernel_launch(void* const* d_in, const int* in_sizes, int n_in,
                              void* d_out, int out_size, void* d_ws, size_t ws_size,
                              hipStream_t stream)
{
    (void)in_sizes; (void)n_in; (void)out_size; (void)ws_size;

    const int*   x    = (const int*)  d_in[0];
    const float* emb  = (const float*)d_in[1];
    const float* w_gx = (const float*)d_in[2];
    const float* w_gh = (const float*)d_in[3];
    const float* b_g  = (const float*)d_in[4];
    const float* w_ix = (const float*)d_in[5];
    const float* w_ih = (const float*)d_in[6];
    const float* b_i  = (const float*)d_in[7];
    const float* w_fx = (const float*)d_in[8];
    const float* w_fh = (const float*)d_in[9];
    const float* b_f  = (const float*)d_in[10];
    const float* w_ox = (const float*)d_in[11];
    const float* w_oh = (const float*)d_in[12];
    const float* b_o  = (const float*)d_in[13];
    const float* w_ph = (const float*)d_in[14];
    const float* b_p  = (const float*)d_in[15];

    // Workspace layout:
    //   whp  : 8 MB   packed wh f16 fragments
    //   xtab : 48 KB  [3][4096] gate LUT (rounded to 64KB)
    //   h_a  : 512 KB f16 ping
    //   h_b  : 512 KB f16 pong
    //   cst  : 1 MB   f32 cell state
    char* ws = (char*)d_ws;
    _Float16* whp  = (_Float16*)ws;
    float*    xtab = (float*)(ws + (size_t)(8u << 20));
    _Float16* h_a  = (_Float16*)(ws + (size_t)(8u << 20) + 65536);
    _Float16* h_b  = h_a + (size_t)BATCH * HID;
    float*    cst  = (float*)(ws + (size_t)(8u << 20) + 65536
                              + 2 * (size_t)BATCH * HID * sizeof(_Float16));

    // h0 = 0, c0 = 0 (capture-safe async memsets)
    hipMemsetAsync(h_a, 0, (size_t)BATCH * HID * sizeof(_Float16), stream);
    hipMemsetAsync(cst, 0, (size_t)BATCH * HID * sizeof(float), stream);

    // One-time (per call) weight pack + gate LUT
    pack_wh_kernel<<<1024, 256, 0, stream>>>(w_gh, w_ih, w_fh, w_oh, whp);
    gate_table_kernel<<<48, 256, 0, stream>>>(emb, w_gx, w_ix, w_fx, w_ox,
                                              b_g, b_i, b_f, b_o, xtab);

    // Sequential recurrence: stream order provides the step barrier.
    for (int t = 0; t < SEQ; ++t) {
        const _Float16* hin  = (t & 1) ? h_b : h_a;
        _Float16*       hout = (t & 1) ? h_a : h_b;
        lstm_step_kernel<<<128, 256, 0, stream>>>(hin, hout, cst, whp, xtab, x, t);
    }

    // Final h lands in h_a after 512 steps (even count of swaps).
    proj_logsoftmax_kernel<<<1, 256, 0, stream>>>(h_a, w_ph, b_p, (float*)d_out);
}
